// KANFeatureFusion_57518202028318
// MI455X (gfx1250) — compile-verified, matrix operand order checked by reading, not run on hardware
//
#include <hip/hip_runtime.h>
#include <hip/hip_bf16.h>

// ---------------------------------------------------------------------------
// KAN feature fusion for MI455X (gfx1250, wave32, WMMA 16x16x32 f16 -> f32)
//   layer1: (32768 x 512) -> 256   base K=512, spline K=4096
//   layer2: (32768 x 256) -> 256   base K=256, spline K=2048
//   layernorm over last dim (256), in-place on d_out
// 32-token M tile per block, 2x2 WMMA subtiles per wave, closed-form
// uniform cubic B-spline basis generation straight into LDS.
// ---------------------------------------------------------------------------

typedef __attribute__((ext_vector_type(16))) _Float16 v16h;
typedef __attribute__((ext_vector_type(8)))  _Float16 v8h;
typedef __attribute__((ext_vector_type(4)))  _Float16 v4h;
typedef __attribute__((ext_vector_type(2)))  _Float16 v2h;
typedef __attribute__((ext_vector_type(8)))  float    v8f;

#define TOKENS  32768
#define OUTF    256
#define NB      8
#define KCHUNK  512       // spline K chunk = 64 input features; 32 KB LDS tile
#define FCHUNK  (KCHUNK / NB)
#define TILE_M  32

__device__ __forceinline__ float silu_f(float x) {
  return x * (1.0f / (1.0f + __expf(-x)));
}

// Closed-form uniform cubic B-spline (knots g[i] = -1 + (i-3)*0.4, i=0..11).
// For x in knot cell i (u = (x - g0)/h), the nonzero cubic bases are
// j = i-3..i with the standard blending weights of t = frac(u); indices
// outside [0,8) are dropped (matches Cox-de Boor truncation at array bounds).
__device__ __forceinline__ void bspline_store(float x, _Float16* __restrict__ slot) {
  v8h z = {};
  *(v8h*)slot = z;                      // zero the 8-basis slot (ds_store_b128)
  float u = (x + 2.2f) * 2.5f;          // g0 = -2.2, 1/h = 2.5
  int i = (int)floorf(u);
  if (i >= 0 && i <= 10) {
    float t   = u - (float)i;
    float omt = 1.0f - t;
    float t2 = t * t, t3 = t2 * t;
    float w0 = omt * omt * omt * (1.0f / 6.0f);                     // j = i-3
    float w1 = (3.0f * t3 - 6.0f * t2 + 4.0f) * (1.0f / 6.0f);      // j = i-2
    float w2 = (-3.0f * t3 + 3.0f * t2 + 3.0f * t + 1.0f) * (1.0f / 6.0f); // j=i-1
    float w3 = t3 * (1.0f / 6.0f);                                  // j = i
    int j = i - 3;
    if (j     >= 0 && j     < 8) slot[j]     = (_Float16)w0;
    if (j + 1 >= 0 && j + 1 < 8) slot[j + 1] = (_Float16)w1;
    if (j + 2 >= 0 && j + 2 < 8) slot[j + 2] = (_Float16)w2;
    if (j + 3 >= 0 && j + 3 < 8) slot[j + 3] = (_Float16)w3;
  }
}

// 16-bit A/B fragment loader per CDNA5 ISA 7.12.2 layout:
//   lane&15 = M (A) or N (B); lane>>4 selects K-half (+8);
//   VGPR v in 0..3 -> K = 2v (+khalf*8); v in 4..7 -> K = 16 + 2(v-4) (+khalf*8)
// K pairs 0..7 and 16..23 are contiguous -> compiler emits two b128 loads.
__device__ __forceinline__ v16h load_frag(const _Float16* __restrict__ p,
                                          int stride, int kbase, int lane) {
  const _Float16* rp = p + (size_t)(lane & 15) * stride + kbase + ((lane >> 4) << 3);
  v16h r;
#pragma unroll
  for (int v = 0; v < 8; ++v) {
    int k = (v < 4) ? (v * 2) : (16 + (v - 4) * 2);
    v2h pr = *(const v2h*)(rp + k);
    r[2 * v]     = pr.x;
    r[2 * v + 1] = pr.y;
  }
  return r;
}

// ---------------------------------------------------------------------------
// Weight prep: f32 -> f16, folding the standalone spline scaler into spline_w.
// ---------------------------------------------------------------------------
__global__ void cvt_base_kernel(const float* __restrict__ W,
                                _Float16* __restrict__ O, int n) {
  int i = blockIdx.x * blockDim.x + threadIdx.x;
  if (i < n) O[i] = (_Float16)W[i];
}

__global__ void cvt_spline_kernel(const float* __restrict__ SW,
                                  const float* __restrict__ SC,
                                  _Float16* __restrict__ O, int n) {
  int i = blockIdx.x * blockDim.x + threadIdx.x;
  if (i < n) O[i] = (_Float16)(SW[i] * SC[i >> 3]);   // SC indexed per (out,in)
}

// ---------------------------------------------------------------------------
// Fused KANLinear layer: base (silu) path + spline path, f16 WMMA, f32 accum.
// Block = 256 threads (8 wave32s) -> 32 tokens x 256 outputs.
// Each wave owns two 16-wide N-tiles x two 16-tall M-tiles (4 accumulators).
// IN = 512 (layer1, concat inputs) or 256 (layer2).
// ---------------------------------------------------------------------------
template <int IN>
__global__ __launch_bounds__(256) void kan_layer(
    const float* __restrict__ X1, const float* __restrict__ X2,
    const _Float16* __restrict__ Wb,   // (256, IN)      f16
    const _Float16* __restrict__ Ws,   // (256, IN*8)    f16 (scaler folded)
    float* __restrict__ Y)             // (TOKENS, 256)  f32
{
  __shared__ _Float16 smA[TILE_M * KCHUNK];   // 32 KB staging tile

  const int tid  = threadIdx.x;
  const int lane = tid & 31;
  const int wave = tid >> 5;
  const int t0   = blockIdx.x * TILE_M;

  const int ob0 = wave * 32;        // first N-tile (16 outputs)
  const int ob1 = ob0 + 16;         // second N-tile

  v8f acc00 = {}, acc01 = {};       // M-tile 0 x {N0, N1}
  v8f acc10 = {}, acc11 = {};       // M-tile 1 x {N0, N1}

  // ---------------- base path: A = silu(x), K = IN (fits one LDS tile) -----
  for (int p = tid; p < TILE_M * (IN / 4); p += 256) {
    int m  = p / (IN / 4);
    int f0 = (p % (IN / 4)) * 4;
    float4 xv;
    if constexpr (IN == 512) {
      int t = t0 + m;
      xv = (f0 < 256) ? ((const float4*)(X1 + (size_t)t * 256))[f0 >> 2]
                      : ((const float4*)(X2 + (size_t)t * 256))[(f0 - 256) >> 2];
    } else {
      xv = ((const float4*)(X1 + (size_t)(t0 + m) * IN))[f0 >> 2];
    }
    v4h o;
    o.x = (_Float16)silu_f(xv.x);
    o.y = (_Float16)silu_f(xv.y);
    o.z = (_Float16)silu_f(xv.z);
    o.w = (_Float16)silu_f(xv.w);
    *(v4h*)(smA + m * IN + f0) = o;
  }
  __syncthreads();

#pragma unroll 2
  for (int ks = 0; ks < IN; ks += 32) {
    v16h b0 = load_frag(Wb + (size_t)ob0 * IN, IN, ks, lane);
    v16h b1 = load_frag(Wb + (size_t)ob1 * IN, IN, ks, lane);
    v16h a0 = load_frag(smA,                 IN, ks, lane);
    v16h a1 = load_frag(smA + 16 * IN,       IN, ks, lane);
    acc00 = __builtin_amdgcn_wmma_f32_16x16x32_f16(false, a0, false, b0, (short)0, acc00, false, false);
    acc01 = __builtin_amdgcn_wmma_f32_16x16x32_f16(false, a0, false, b1, (short)0, acc01, false, false);
    acc10 = __builtin_amdgcn_wmma_f32_16x16x32_f16(false, a1, false, b0, (short)0, acc10, false, false);
    acc11 = __builtin_amdgcn_wmma_f32_16x16x32_f16(false, a1, false, b1, (short)0, acc11, false, false);
  }
  __syncthreads();

  // ---------------- spline path: A = bsplines(x), K = IN*8 -----------------
  const int KS = IN * NB;
  for (int c = 0; c < KS; c += KCHUNK) {
    // stage FCHUNK features x 8 bases for 32 tokens into LDS
    for (int p = tid; p < TILE_M * FCHUNK; p += 256) {
      int m  = p / FCHUNK;
      int fi = p % FCHUNK;
      int f  = (c >> 3) + fi;        // absolute input feature index
      float x;
      if constexpr (IN == 512) {
        int t = t0 + m;
        x = (f < 256) ? X1[(size_t)t * 256 + f] : X2[(size_t)t * 256 + (f - 256)];
      } else {
        x = X1[(size_t)(t0 + m) * IN + f];
      }
      bspline_store(x, smA + m * KCHUNK + fi * NB);
    }
    __syncthreads();

    // hint the next weight chunk toward the caches (global_prefetch_b8)
    __builtin_prefetch(Ws + (size_t)ob0 * KS + c + KCHUNK, 0, 0);

#pragma unroll 2
    for (int ks = 0; ks < KCHUNK; ks += 32) {
      v16h b0 = load_frag(Ws + (size_t)ob0 * KS + c, KS, ks, lane);
      v16h b1 = load_frag(Ws + (size_t)ob1 * KS + c, KS, ks, lane);
      v16h a0 = load_frag(smA,               KCHUNK, ks, lane);
      v16h a1 = load_frag(smA + 16 * KCHUNK, KCHUNK, ks, lane);
      acc00 = __builtin_amdgcn_wmma_f32_16x16x32_f16(false, a0, false, b0, (short)0, acc00, false, false);
      acc01 = __builtin_amdgcn_wmma_f32_16x16x32_f16(false, a0, false, b1, (short)0, acc01, false, false);
      acc10 = __builtin_amdgcn_wmma_f32_16x16x32_f16(false, a1, false, b0, (short)0, acc10, false, false);
      acc11 = __builtin_amdgcn_wmma_f32_16x16x32_f16(false, a1, false, b1, (short)0, acc11, false, false);
    }
    __syncthreads();
  }

  // ------- store D (f32 16x16 layout: VGPR r -> M = r + 8*(lane>>4)) -------
  const int col = lane & 15;
  const int mb  = (lane >> 4) * 8;
#pragma unroll
  for (int r = 0; r < 8; ++r) {
    int ta = t0 + mb + r;
    int tb = ta + 16;
    Y[(size_t)ta * OUTF + ob0 + col] = acc00[r];
    Y[(size_t)ta * OUTF + ob1 + col] = acc01[r];
    Y[(size_t)tb * OUTF + ob0 + col] = acc10[r];
    Y[(size_t)tb * OUTF + ob1 + col] = acc11[r];
  }
}

// ---------------------------------------------------------------------------
// LayerNorm over 256 features; one wave32 per row, in-place.
// ---------------------------------------------------------------------------
__global__ __launch_bounds__(256) void layernorm_kernel(
    float* __restrict__ Y, const float* __restrict__ w, const float* __restrict__ b)
{
  const int lane = threadIdx.x & 31;
  const int wave = threadIdx.x >> 5;
  const int row  = blockIdx.x * 8 + wave;

  float v[8];
  float s = 0.0f;
#pragma unroll
  for (int i = 0; i < 8; ++i) {
    v[i] = Y[(size_t)row * 256 + i * 32 + lane];
    s += v[i];
  }
#pragma unroll
  for (int off = 16; off > 0; off >>= 1) s += __shfl_xor(s, off, 32);
  float mu = s * (1.0f / 256.0f);

  float q = 0.0f;
#pragma unroll
  for (int i = 0; i < 8; ++i) { float d = v[i] - mu; q += d * d; }
#pragma unroll
  for (int off = 16; off > 0; off >>= 1) q += __shfl_xor(q, off, 32);
  float rstd = rsqrtf(q * (1.0f / 256.0f) + 1e-5f);

#pragma unroll
  for (int i = 0; i < 8; ++i) {
    int c = i * 32 + lane;
    Y[(size_t)row * 256 + c] = (v[i] - mu) * rstd * w[c] + b[c];
  }
}

// ---------------------------------------------------------------------------
extern "C" void kernel_launch(void* const* d_in, const int* in_sizes, int n_in,
                              void* d_out, int out_size, void* d_ws, size_t ws_size,
                              hipStream_t stream) {
  const float* temporal  = (const float*)d_in[0];  // (16,2048,256)
  const float* spatial   = (const float*)d_in[1];  // (16,2048,256)
  const float* base_w1   = (const float*)d_in[2];  // (256,512)
  const float* spline_w1 = (const float*)d_in[3];  // (256,512,8)
  const float* scaler1   = (const float*)d_in[4];  // (256,512)
  const float* base_w2   = (const float*)d_in[5];  // (256,256)
  const float* spline_w2 = (const float*)d_in[6];  // (256,256,8)
  const float* scaler2   = (const float*)d_in[7];  // (256,256)
  const float* ln_w      = (const float*)d_in[8];  // (256,)
  const float* ln_b      = (const float*)d_in[9];  // (256,)
  float* out = (float*)d_out;                      // (16,2048,256) f32

  // workspace carve-up (~35.4 MB total)
  char* ws = (char*)d_ws;
  float*    Hbuf = (float*)ws;      ws += (size_t)TOKENS * 256 * sizeof(float);
  _Float16* Wb1  = (_Float16*)ws;   ws += (size_t)256 * 512 * 2;
  _Float16* Ws1  = (_Float16*)ws;   ws += (size_t)256 * 4096 * 2;
  _Float16* Wb2  = (_Float16*)ws;   ws += (size_t)256 * 256 * 2;
  _Float16* Ws2  = (_Float16*)ws;

  (void)in_sizes; (void)n_in; (void)out_size; (void)ws_size;

  // weight prep (f32 -> f16, fold spline scaler)
  cvt_base_kernel  <<<(131072  + 255) / 256, 256, 0, stream>>>(base_w1, Wb1, 131072);
  cvt_base_kernel  <<<(65536   + 255) / 256, 256, 0, stream>>>(base_w2, Wb2, 65536);
  cvt_spline_kernel<<<(1048576 + 255) / 256, 256, 0, stream>>>(spline_w1, scaler1, Ws1, 1048576);
  cvt_spline_kernel<<<(524288  + 255) / 256, 256, 0, stream>>>(spline_w2, scaler2, Ws2, 524288);

  // layer 1: concat(temporal, spatial) -> h
  kan_layer<512><<<TOKENS / TILE_M, 256, 0, stream>>>(temporal, spatial, Wb1, Ws1, Hbuf);
  // layer 2: h -> y (written straight into d_out)
  kan_layer<256><<<TOKENS / TILE_M, 256, 0, stream>>>(Hbuf, nullptr, Wb2, Ws2, out);
  // layernorm in-place
  layernorm_kernel<<<TOKENS / 8, 256, 0, stream>>>(out, ln_w, ln_b);
}
